// AttentionKernel_24043226923186
// MI455X (gfx1250) — matrix operand order repaired and checked by reading, not compiled, not run on hardware
//
#include <hip/hip_runtime.h>
#include <hip/hip_bf16.h>

// Flash attention (B=2,S=2048,H=16,C=64) + dense bias (B,H,S,S), causal +
// segment masking (m sorted). Transposed-logits formulation: S^T = K Q^T so
// softmax reduces in-lane; P^T registers ARE the A-matrix for the PV WMMA.
// V tiles staged by the Tensor Data Mover (TENSORcnt). Segment lower bounds
// found by binary search (m sorted) -> fully-masked key blocks are skipped.

typedef __bf16 bf16_t;
typedef __attribute__((ext_vector_type(16))) bf16_t v16bf;
typedef __attribute__((ext_vector_type(8)))  float  v8f;
typedef __attribute__((ext_vector_type(4)))  unsigned int   u32x4;
typedef __attribute__((ext_vector_type(8)))  int            i32x8;
typedef __attribute__((ext_vector_type(4)))  int            i32x4;

#define ATT_B 2
#define ATT_S 2048
#define ATT_H 16
#define ATT_C 64
#define WPB   4          // waves per block (128 threads)
#define SCALE 0.125f     // 1/sqrt(64)
#define BPAD  33         // padded bias-tile row stride (bank-conflict free)

__global__ __launch_bounds__(128) void
fa_fwd_kernel(const float* __restrict__ q, const float* __restrict__ k,
              const float* __restrict__ v, const float* __restrict__ bias,
              const int* __restrict__ seg, float* __restrict__ out)
{
    __shared__ __align__(16) float lds_b[WPB][16 * BPAD];  // bias tile [q][key]
    __shared__ __align__(16) float lds_v[WPB][32 * 64];    // V tile f32 [key][ch] (TDM dest)

    const int lane = threadIdx.x & 31;
    const int wid  = threadIdx.x >> 5;
    const int half = lane >> 4;
    const int l16  = lane & 15;

    const int gw    = blockIdx.x * WPB + wid;       // 0..4095
    const int qtile = gw & 127;
    const int bh    = gw >> 7;
    const int b     = bh >> 4;
    const int h     = bh & 15;
    const int q0    = qtile * 16;
    const int qme   = q0 + l16;                     // this lane's query

    // ---- Q tile -> two bf16 B-matrices (B layout: lane=query col,
    //      element i -> channel half*16+i within 32-chunk) ----
    v16bf bq[2];
    {
        const float* qp = q + (((size_t)(b * ATT_S + qme)) * ATT_H + h) * ATT_C;
#pragma unroll
        for (int cc = 0; cc < 2; ++cc)
#pragma unroll
            for (int i = 0; i < 16; ++i)
                bq[cc][i] = (bf16_t)qp[cc * 32 + half * 16 + i];
    }

    // ---- segment lower bound per query (m is sorted) ----
    const int* mb = seg + b * ATT_S;
    const int seg_q = mb[qme];
    int lo = 0, hi = ATT_S;
    while (lo < hi) { int mid = (lo + hi) >> 1; if (mb[mid] < seg_q) lo = mid + 1; else hi = mid; }
    const int klo = lo;                 // first key in my query's segment
    int mn = klo;
#pragma unroll
    for (int m = 1; m <= 16; m <<= 1) mn = min(mn, __shfl_xor(mn, m, 32));
    const int jstart = __builtin_amdgcn_readfirstlane(mn) >> 5;
    // klo broadcast per O-row (query r+8*half), loop-invariant
    int kloR[8];
#pragma unroll
    for (int r = 0; r < 8; ++r) kloR[r] = __shfl(klo, r + 8 * half, 32);

    float rm = -1e30f, rl = 0.0f;       // per-lane (per-query) running stats
    v8f o[4];
#pragma unroll
    for (int ct = 0; ct < 4; ++ct)
#pragma unroll
        for (int r = 0; r < 8; ++r) o[ct][r] = 0.0f;

    // ---- constant parts of the TDM descriptor (ISA 08 §8.3/§8.4) ----
    const unsigned lds_v_addr = (unsigned)__builtin_amdgcn_readfirstlane(
        (int)(unsigned)(uintptr_t)(void*)&lds_v[wid][0]);
    i32x8 g1;
    g1[0] = 0x20000;            // data_size=2 (4B)
    g1[1] = (64 << 16);         // tensor_dim0 = 64
    g1[2] = (32 << 16);         // tensor_dim1 = 32
    g1[3] = (64 << 16);         // tile_dim0 = 64
    g1[4] = 32;                 // tile_dim1 = 32
    g1[5] = ATT_H * ATT_C;      // tensor_dim0_stride = 1024 elements
    g1[6] = 0;
    g1[7] = 0;
    const i32x4 gz4 = {0, 0, 0, 0};
    const i32x8 gz8 = {0, 0, 0, 0, 0, 0, 0, 0};

    const int nblocks = q0 / 32 + 1;     // causal: keys [0, q0+16)
    for (int j = jstart; j < nblocks; ++j) {
        const int k0 = j * 32;

        // previous iteration's LDS reads must retire before TDM rewrites lds_v
        asm volatile("s_wait_dscnt 0" ::: "memory");

        // ---- async V tile DMA: global (32 keys x 64 ch f32) -> LDS ----
        {
            const unsigned long long ga = (unsigned long long)(uintptr_t)(const void*)
                (v + (((size_t)(b * ATT_S + k0)) * ATT_H + h) * ATT_C);
            u32x4 g0;
            g0[0] = 1u;
            g0[1] = lds_v_addr;
            g0[2] = (unsigned)__builtin_amdgcn_readfirstlane((int)(unsigned)ga);
            g0[3] = (unsigned)__builtin_amdgcn_readfirstlane(
                        (int)((unsigned)((ga >> 32) & 0x01FFFFFFull) | (2u << 30)));
            __builtin_amdgcn_tensor_load_to_lds(g0, g1, gz4, gz4, gz8, 0);
        }

        // ---- transposed logits: S^T(t) = K(t) Q^T, t in {0,1} ----
        v8f st0, st1;
#pragma unroll
        for (int r = 0; r < 8; ++r) { st0[r] = 0.0f; st1[r] = 0.0f; }
#pragma unroll
        for (int cc = 0; cc < 2; ++cc) {
            v16bf ak0, ak1;   // A layout: lane=key row, elem i -> ch half*16+i
            const float* kp0 = k + (((size_t)(b * ATT_S + k0 + l16)) * ATT_H + h) * ATT_C
                               + cc * 32;
            const float* kp1 = kp0 + (size_t)16 * ATT_H * ATT_C;
#pragma unroll
            for (int i = 0; i < 16; ++i) {
                int kc = half * 16 + i;
                ak0[i] = (bf16_t)kp0[kc];
                ak1[i] = (bf16_t)kp1[kc];
            }
            st0 = __builtin_amdgcn_wmma_f32_16x16x32_bf16(false, ak0, false, bq[cc],
                                                          (short)0, st0, false, false);
            st1 = __builtin_amdgcn_wmma_f32_16x16x32_bf16(false, ak1, false, bq[cc],
                                                          (short)0, st1, false, false);
        }

        // ---- stage bias tile to LDS transposed (global loads stay coalesced;
        //      loads predicated by exact mask -> masked lines never fetched) ----
#pragma unroll
        for (int r = 0; r < 8; ++r) {
            const int qw = q0 + r + 8 * half;       // writer-orientation query row
            const float* brow = bias + (((size_t)(b * ATT_H + h)) * ATT_S + qw) * ATT_S + k0;
            const bool w0 = (k0 + l16 >= kloR[r]) && (k0 + l16 <= qw);
            const bool w1 = (k0 + 16 + l16 >= kloR[r]) && (k0 + 16 + l16 <= qw);
            lds_b[wid][(r + 8 * half) * BPAD + l16]      = w0 ? brow[l16]      : 0.0f;
            lds_b[wid][(r + 8 * half) * BPAD + 16 + l16] = w1 ? brow[16 + l16] : 0.0f;
            if (j + 1 < nblocks) __builtin_prefetch(brow + 32 + l16, 0, 3);
        }
        asm volatile("s_wait_dscnt 0" ::: "memory");

        // ---- in-lane online softmax over this lane's 16 keys ----
        float v0[8], v1[8];
        float bm = -1e30f;
#pragma unroll
        for (int r = 0; r < 8; ++r) {
            const int kk0 = k0 + r + 8 * half;
            const int kk1 = kk0 + 16;
            const float b0 = lds_b[wid][l16 * BPAD + r + 8 * half];
            const float b1 = lds_b[wid][l16 * BPAD + 16 + r + 8 * half];
            const bool ok0 = (kk0 >= klo) && (kk0 <= qme);
            const bool ok1 = (kk1 >= klo) && (kk1 <= qme);
            v0[r] = ok0 ? (st0[r] * SCALE + b0) : -1e30f;
            v1[r] = ok1 ? (st1[r] * SCALE + b1) : -1e30f;
            bm = fmaxf(bm, fmaxf(v0[r], v1[r]));
        }
        bm = fmaxf(bm, __shfl_xor(bm, 16, 32));     // other half of the keys
        const float nm = fmaxf(rm, bm);
        const float alpha = __expf(rm - nm);
        rm = nm;

        // P^T registers are exactly the PV A-matrix layout: pack directly
        v16bf ap;
        float bs = 0.0f;
#pragma unroll
        for (int r = 0; r < 8; ++r) {
            const float e0 = (v0[r] > -1e30f) ? __expf(v0[r] - nm) : 0.0f;
            const float e1 = (v1[r] > -1e30f) ? __expf(v1[r] - nm) : 0.0f;
            bs += e0 + e1;
            ap[r]     = (bf16_t)e0;
            ap[r + 8] = (bf16_t)e1;
        }
        bs += __shfl_xor(bs, 16, 32);
        rl = rl * alpha + bs;

        // rescale O (normal orientation: VGPR r = query r+8*half) by alpha
#pragma unroll
        for (int r = 0; r < 8; ++r) {
            const float ar = __shfl(alpha, r + 8 * half, 32);
#pragma unroll
            for (int ct = 0; ct < 4; ++ct) o[ct][r] *= ar;
        }

        // ---- V tile ready -> O += P V ----
        __builtin_amdgcn_s_wait_tensorcnt(0);
#pragma unroll
        for (int ct = 0; ct < 4; ++ct) {
            v16bf bv;   // B layout: lane=channel col, elem i -> key (i<8 ? half*8+i : 16+half*8+i-8)
#pragma unroll
            for (int i = 0; i < 16; ++i)
                bv[i] = (bf16_t)lds_v[wid][((i < 8) ? (half * 8 + i) : (16 + half * 8 + (i - 8))) * 64 + ct * 16 + l16];
            o[ct] = __builtin_amdgcn_wmma_f32_16x16x32_bf16(false, ap, false, bv,
                                                            (short)0, o[ct], false, false);
        }
    }

    // ---- epilogue: normalize and store (coalesced 64B rows) ----
#pragma unroll
    for (int r = 0; r < 8; ++r) {
        const float lr = __shfl(rl, r + 8 * half, 32);
        const float il = (lr > 0.0f) ? (1.0f / lr) : 0.0f;
        const int row = q0 + r + 8 * half;
        float* op = out + (((size_t)(b * ATT_S + row)) * ATT_H + h) * ATT_C + l16;
#pragma unroll
        for (int ct = 0; ct < 4; ++ct) op[ct * 16] = o[ct][r] * il;
    }
}

extern "C" void kernel_launch(void* const* d_in, const int* in_sizes, int n_in,
                              void* d_out, int out_size, void* d_ws, size_t ws_size,
                              hipStream_t stream) {
    (void)in_sizes; (void)n_in; (void)out_size; (void)d_ws; (void)ws_size;
    const float* q    = (const float*)d_in[0];
    const float* k    = (const float*)d_in[1];
    const float* v    = (const float*)d_in[2];
    const float* bias = (const float*)d_in[3];
    const int*   seg  = (const int*)d_in[4];
    float* out        = (float*)d_out;

    dim3 grid(ATT_B * ATT_H * (ATT_S / 16) / WPB);   // 1024 blocks
    dim3 block(WPB * 32);                            // 128 threads = 4 waves
    fa_fwd_kernel<<<grid, block, 0, stream>>>(q, k, v, bias, seg, out);
}
